// TemporalEGCN_2783138808296
// MI455X (gfx1250) — matrix operand order repaired and checked by reading, not compiled, not run on hardware
//
#include <hip/hip_runtime.h>

// MI455X / gfx1250, wave32. Matrix math via V_WMMA_F32_16X16X32_F16.
// Heavy GEMMs use double-buffered global_load_async_to_lds_b128 pipelines:
// issue next K-block (3 async b128/thread) -> s_wait_asynccnt 0x3 (previous
// group landed, new group still in flight) -> barrier -> WMMA on current buf.
typedef __attribute__((ext_vector_type(16))) _Float16 v16h;
typedef __attribute__((ext_vector_type(8)))  _Float16 h8;
typedef __attribute__((ext_vector_type(8)))  float    v8f;

#define WMMA(a, b, c) \
  __builtin_amdgcn_wmma_f32_16x16x32_f16(false, (a), false, (b), (short)0, (c), false, false)

static const int    kBN = 8 * 512;                    // 4096
static const size_t kE  = (size_t)8 * 4 * 512 * 512;  // 8388608 (B*P*N*N)

// LDS row pitches chosen so every 8-half chunk is 16B aligned (b128-able)
#define LDA_PAD 40   // A tiles: 128 x 32 halves, pitch 40 (80B rows)
#define LDB_PAD 72   // B tiles: 32 x 64 halves, pitch 72 (144B rows)

// ---------------- WMMA fragment loaders (LDS -> VGPR, k = 16*(lane/16) + s for A and B) -------

__device__ static inline v16h load_a_frag(const _Float16* s, int ld) {
  // s: 16 x (>=32) row-major tile, element (m,k) at s[m*ld+k]
  int lane = threadIdx.x & 31;
  int m = lane & 15, hh = lane >> 4;
  v16h a;
#pragma unroll
  for (int v = 0; v < 8; ++v) {
    int k = hh * 16 + 2 * v;
    a[2 * v]     = s[m * ld + k];
    a[2 * v + 1] = s[m * ld + k + 1];
  }
  return a;
}

__device__ static inline v16h load_b_frag(const _Float16* s, int ld) {
  // s: 32 x (>=16) row-major K x N tile, element (k,n) at s[k*ld+n]
  int lane = threadIdx.x & 31;
  int n = lane & 15, hh = lane >> 4;
  v16h b;
#pragma unroll
  for (int v = 0; v < 8; ++v) {
    int k = hh * 16 + 2 * v;
    b[2 * v]     = s[k * ld + n];
    b[2 * v + 1] = s[(k + 1) * ld + n];
  }
  return b;
}

__device__ static inline v16h load_bT_frag(const _Float16* s, int ld) {
  // s: 16 x (>=32) row-major N x K tile; B[k][n] = s[n*ld+k]  (A @ A^T)
  int lane = threadIdx.x & 31;
  int n = lane & 15, hh = lane >> 4;
  v16h b;
#pragma unroll
  for (int v = 0; v < 8; ++v) {
    int k = hh * 16 + 2 * v;
    b[2 * v]     = s[n * ld + k];
    b[2 * v + 1] = s[n * ld + k + 1];
  }
  return b;
}

// 8 consecutive f32 -> packed 8 x f16 (two global_load_b128 + v_cvt)
__device__ static inline h8 cvt8(const float* g) {
  float4 f0 = *(const float4*)g;
  float4 f1 = *(const float4*)(g + 4);
  h8 r;
  r[0] = (_Float16)f0.x; r[1] = (_Float16)f0.y;
  r[2] = (_Float16)f0.z; r[3] = (_Float16)f0.w;
  r[4] = (_Float16)f1.x; r[5] = (_Float16)f1.y;
  r[6] = (_Float16)f1.z; r[7] = (_Float16)f1.w;
  return r;
}

// Generic LDS address -> 32-bit LDS offset (low 32 bits of the address)
__device__ static inline unsigned ldsoff(const void* p) {
  return (unsigned)(size_t)p;
}

// Issue one K-block stage: 3 x 16B async copies per thread
__device__ static inline void async_issue3(unsigned l0, const _Float16* g0,
                                           unsigned l1, const _Float16* g1,
                                           unsigned l2, const _Float16* g2) {
  asm volatile("global_load_async_to_lds_b128 %0, %1, off" :: "v"(l0), "v"(g0) : "memory");
  asm volatile("global_load_async_to_lds_b128 %0, %1, off" :: "v"(l1), "v"(g1) : "memory");
  asm volatile("global_load_async_to_lds_b128 %0, %1, off" :: "v"(l2), "v"(g2) : "memory");
}

// ---------------- Sinkhorn normalization (_ds) -------------------------------

// rsinv[b,p,i] = 1 / sum_k e[b,i,k,p] ; one wave32 per row
__global__ void k_rowsum(const float* __restrict__ e, float* __restrict__ rsinv) {
  int wid  = (blockIdx.x << 3) + (threadIdx.x >> 5);
  int lane = threadIdx.x & 31;
  int i = wid & 511, p = (wid >> 9) & 3, b = wid >> 11;
  const float* base = e + (((size_t)b * 512 + i) * 512) * 4 + p;
  float s = 0.f;
  for (int k = lane; k < 512; k += 32) s += base[(size_t)k * 4];
#pragma unroll
  for (int off = 16; off > 0; off >>= 1) s += __shfl_down(s, off, 32);
  if (lane == 0) rsinv[wid] = 1.f / s;
}

// csinv[b,p,k] = 1 / sum_i e[b,i,k,p]*rsinv[b,p,i]
__global__ void k_colsum(const float* __restrict__ e, const float* __restrict__ rsinv,
                         float* __restrict__ csinv) {
  int t = blockIdx.x * 256 + threadIdx.x;
  int k = t & 511, p = (t >> 9) & 3, b = t >> 11;
  const float* base = e + (size_t)b * 512 * 512 * 4 + (size_t)k * 4 + p;
  const float* r = rsinv + (b * 4 + p) * 512;
  float s = 0.f;
  for (int i = 0; i < 512; ++i) s += base[(size_t)i * 2048] * r[i];
  csinv[t] = 1.f / s;
}

// a2h[(b*P+p)*N*N + i*N + k] = f16( e[b,i,k,p] * rsinv * csinv )
__global__ void k_a2h(const float* __restrict__ e, const float* __restrict__ rsinv,
                      const float* __restrict__ csinv, _Float16* __restrict__ a2h) {
  size_t t = (size_t)blockIdx.x * 256 + threadIdx.x;
  int k = (int)(t & 511), i = (int)(t >> 9) & 511, p = (int)(t >> 18) & 3, b = (int)(t >> 20);
  float v = e[(((size_t)b * 512 + i) * 512 + k) * 4 + p] *
            rsinv[(b * 4 + p) * 512 + i] * csinv[(b * 4 + p) * 512 + k];
  a2h[t] = (_Float16)v;
}

// m = a2 @ a2^T per (b,p); e_out[b,i,j,p] = m[i,j].
// 128x64 block tile, 8 waves x (2x2) sub-tiles, double-buffered async staging.
// grid(32, 32), block 256
__global__ __launch_bounds__(256) void k_ds_gemm(const _Float16* __restrict__ a2h,
                                                 float* __restrict__ e_out) {
  __shared__ __align__(16) _Float16 As[2][128][LDA_PAD];
  __shared__ __align__(16) _Float16 Bs[2][64][LDA_PAD];
  int bp = blockIdx.y;
  const _Float16* base = a2h + (size_t)bp * 512 * 512;
  int ti = (blockIdx.x >> 3) * 128;   // i tile (4)
  int tj = (blockIdx.x & 7) * 64;     // j tile (8)
  int t = threadIdx.x, w = t >> 5, lane = t & 31;
  int mi = (w >> 1) * 32, ni = (w & 1) * 32;
  int rA = t >> 2, cA = (t & 3) * 8;  // A: 2 chunks/thread; B: 1 chunk/thread
  unsigned lA0_0 = ldsoff(&As[0][rA][cA]),      lA0_1 = ldsoff(&As[1][rA][cA]);
  unsigned lA1_0 = ldsoff(&As[0][rA + 64][cA]), lA1_1 = ldsoff(&As[1][rA + 64][cA]);
  unsigned lB_0  = ldsoff(&Bs[0][rA][cA]),      lB_1  = ldsoff(&Bs[1][rA][cA]);
  const _Float16* gA0 = &base[(size_t)(ti + rA) * 512 + cA];
  const _Float16* gA1 = &base[(size_t)(ti + rA + 64) * 512 + cA];
  const _Float16* gB  = &base[(size_t)(tj + rA) * 512 + cA];
  v8f acc00 = {}, acc01 = {}, acc10 = {}, acc11 = {};

  async_issue3(lA0_0, gA0, lA1_0, gA1, lB_0, gB);   // prologue: K-block 0 -> buf 0
  for (int k0 = 0; k0 < 512; k0 += 64) {
    // ---- phase 0: compute buf0 (k0), prefetch k0+32 -> buf1
    if (k0 + 32 < 512) {
      async_issue3(lA0_1, gA0 + k0 + 32, lA1_1, gA1 + k0 + 32, lB_1, gB + k0 + 32);
      asm volatile("s_wait_asynccnt 0x3" ::: "memory");
    } else {
      asm volatile("s_wait_asynccnt 0x0" ::: "memory");
    }
    __syncthreads();
    {
      v16h a0 = load_a_frag(&As[0][mi][0], LDA_PAD);
      v16h a1 = load_a_frag(&As[0][mi + 16][0], LDA_PAD);
      v16h b0 = load_bT_frag(&Bs[0][ni][0], LDA_PAD);
      v16h b1 = load_bT_frag(&Bs[0][ni + 16][0], LDA_PAD);
      acc00 = WMMA(a0, b0, acc00);
      acc01 = WMMA(a0, b1, acc01);
      acc10 = WMMA(a1, b0, acc10);
      acc11 = WMMA(a1, b1, acc11);
    }
    __syncthreads();
    // ---- phase 1: compute buf1 (k0+32), prefetch k0+64 -> buf0
    if (k0 + 64 < 512) {
      async_issue3(lA0_0, gA0 + k0 + 64, lA1_0, gA1 + k0 + 64, lB_0, gB + k0 + 64);
      asm volatile("s_wait_asynccnt 0x3" ::: "memory");
    } else {
      asm volatile("s_wait_asynccnt 0x0" ::: "memory");
    }
    __syncthreads();
    {
      v16h a0 = load_a_frag(&As[1][mi][0], LDA_PAD);
      v16h a1 = load_a_frag(&As[1][mi + 16][0], LDA_PAD);
      v16h b0 = load_bT_frag(&Bs[1][ni][0], LDA_PAD);
      v16h b1 = load_bT_frag(&Bs[1][ni + 16][0], LDA_PAD);
      acc00 = WMMA(a0, b0, acc00);
      acc01 = WMMA(a0, b1, acc01);
      acc10 = WMMA(a1, b0, acc10);
      acc11 = WMMA(a1, b1, acc11);
    }
    __syncthreads();
  }
  int bb = bp >> 2, p = bp & 3;
  int hh = lane >> 4, nn2 = lane & 15;
  int row0 = ti + mi + 8 * hh;
  int col0 = tj + ni + nn2;
#pragma unroll
  for (int r = 0; r < 8; ++r) {
    size_t o0 = ((size_t)bb * 512 + (row0 + r)) * 512;
    size_t o1 = ((size_t)bb * 512 + (row0 + 16 + r)) * 512;
    e_out[(o0 + col0) * 4 + p]      = acc00[r];
    e_out[(o0 + col0 + 16) * 4 + p] = acc01[r];
    e_out[(o1 + col0) * 4 + p]      = acc10[r];
    e_out[(o1 + col0 + 16) * 4 + p] = acc11[r];
  }
}

// ---------------- Generic f32-in GEMM: C = act(A@B + bias) -------------------
// A[M,K] lda, B[K,N] ldb row-major; dims multiples of 128/64/32 here.
// act: 0 none, 1 sigmoid, 2 tanh.  grid(M/128, N/64), block 256
__global__ __launch_bounds__(256) void k_gemm_f32(
    const float* __restrict__ A, int lda, const float* __restrict__ Bw, int ldb,
    const float* __restrict__ bias, float* __restrict__ C, int ldc, int K, int act) {
  __shared__ __align__(16) _Float16 As[128][LDA_PAD];
  __shared__ __align__(16) _Float16 Bs[32][LDB_PAD];
  int tile_m = blockIdx.x * 128;
  int tile_n = blockIdx.y * 64;
  int t = threadIdx.x, w = t >> 5, lane = t & 31;
  int mi = (w >> 1) * 32, ni = (w & 1) * 32;
  int rA = t >> 2, cA = (t & 3) * 8;
  int rB = t >> 3, cB = (t & 7) * 8;
  v8f acc00 = {}, acc01 = {}, acc10 = {}, acc11 = {};
  for (int k0 = 0; k0 < K; k0 += 32) {
    *(h8*)&As[rA][cA]      = cvt8(&A[(size_t)(tile_m + rA) * lda + k0 + cA]);
    *(h8*)&As[rA + 64][cA] = cvt8(&A[(size_t)(tile_m + rA + 64) * lda + k0 + cA]);
    *(h8*)&Bs[rB][cB]      = cvt8(&Bw[(size_t)(k0 + rB) * ldb + tile_n + cB]);
    __syncthreads();
    v16h a0 = load_a_frag(&As[mi][0], LDA_PAD);
    v16h a1 = load_a_frag(&As[mi + 16][0], LDA_PAD);
    v16h b0 = load_b_frag(&Bs[0][ni], LDB_PAD);
    v16h b1 = load_b_frag(&Bs[0][ni + 16], LDB_PAD);
    acc00 = WMMA(a0, b0, acc00);
    acc01 = WMMA(a0, b1, acc01);
    acc10 = WMMA(a1, b0, acc10);
    acc11 = WMMA(a1, b1, acc11);
    __syncthreads();
  }
  int hh = lane >> 4, nn2 = lane & 15;
  int row0 = tile_m + mi + 8 * hh;
  int cn0  = tile_n + ni + nn2;
  float bv0 = bias[cn0], bv1 = bias[cn0 + 16];
#pragma unroll
  for (int r = 0; r < 8; ++r) {
    float v00 = acc00[r] + bv0, v01 = acc01[r] + bv1;
    float v10 = acc10[r] + bv0, v11 = acc11[r] + bv1;
    if (act == 1) {
      v00 = 1.f / (1.f + __expf(-v00)); v01 = 1.f / (1.f + __expf(-v01));
      v10 = 1.f / (1.f + __expf(-v10)); v11 = 1.f / (1.f + __expf(-v11));
    } else if (act == 2) {
      v00 = tanhf(v00); v01 = tanhf(v01);
      v10 = tanhf(v10); v11 = tanhf(v11);
    }
    C[(size_t)(row0 + r) * ldc + cn0]           = v00;
    C[(size_t)(row0 + r) * ldc + cn0 + 16]      = v01;
    C[(size_t)(row0 + 16 + r) * ldc + cn0]      = v10;
    C[(size_t)(row0 + 16 + r) * ldc + cn0 + 16] = v11;
  }
}

// ---------------- Attention-coefficient path --------------------------------

// Fold linear4 into the x-projections: wa2[p,f] = sum_h w2[p,f,h]*w4a[p,h], etc.
__global__ void k_fold(const float* __restrict__ w2, const float* __restrict__ w4a,
                       const float* __restrict__ b2, const float* __restrict__ w3,
                       const float* __restrict__ w4b, const float* __restrict__ b3,
                       float* __restrict__ wa2, float* __restrict__ wb3,
                       float* __restrict__ cab, int fin) {
  int p = blockIdx.x, f = threadIdx.x;
  if (f < fin) {
    float sa = 0.f, sb = 0.f;
    for (int h = 0; h < 128; ++h) {
      sa += w2[((size_t)p * fin + f) * 128 + h] * w4a[p * 128 + h];
      sb += w3[((size_t)p * fin + f) * 128 + h] * w4b[p * 128 + h];
    }
    wa2[p * fin + f] = sa;
    wb3[p * fin + f] = sb;
  }
  if (f == 0) {
    float ca = 0.f, cb = 0.f;
    for (int h = 0; h < 128; ++h) {
      ca += b2[p * 128 + h] * w4a[p * 128 + h];
      cb += b3[p * 128 + h] * w4b[p * 128 + h];
    }
    cab[p * 2] = ca;
    cab[p * 2 + 1] = cb;
  }
}

// ua[p,b,n] = x[b,n,:]·wa2[p] + b2[p]·w4a[p]; vb likewise
__global__ void k_uavb(const float* __restrict__ x, int fin,
                       const float* __restrict__ wa2, const float* __restrict__ wb3,
                       const float* __restrict__ cab,
                       float* __restrict__ ua, float* __restrict__ vb) {
  int t = blockIdx.x * 256 + threadIdx.x;
  int n = t & 511, b = (t >> 9) & 7, p = t >> 12;
  const float* xv = x + ((size_t)b * 512 + n) * fin;
  const float* wa = wa2 + p * fin;
  const float* wb = wb3 + p * fin;
  float sa = cab[p * 2], sb = cab[p * 2 + 1];
  for (int f = 0; f < fin; ++f) { float xf = xv[f]; sa += xf * wa[f]; sb += xf * wb[f]; }
  ua[t] = sa;
  vb[t] = sb;
}

// coef[p,b,i,j] = sigmoid(leaky_relu(ua[p,b,j] + vb[p,b,i] + b4[p]))
__global__ void k_coef(const float* __restrict__ ua, const float* __restrict__ vb,
                       const float* __restrict__ b4, const float* __restrict__ e_in,
                       _Float16* __restrict__ coefh, float* __restrict__ ne, int write_ne) {
  size_t t = (size_t)blockIdx.x * 256 + threadIdx.x;
  int j = (int)(t & 511), i = (int)(t >> 9) & 511, p = (int)(t >> 18) & 3, b = (int)(t >> 20);
  int pb = (p * 8 + b) * 512;
  float s = ua[pb + j] + vb[pb + i] + b4[p];
  float l = s > 0.f ? s : 0.01f * s;
  float c = 1.f / (1.f + __expf(-l));
  coefh[t] = (_Float16)c;
  if (write_ne) {
    size_t ei = (((size_t)b * 512 + i) * 512 + j) * 4 + p;
    ne[ei] = c * e_in[ei];
  }
}

__global__ void k_f2h(const float* __restrict__ in, _Float16* __restrict__ out) {
  int t = blockIdx.x * 256 + threadIdx.x;
  out[t] = (_Float16)in[t];
}

// xi[b, i, p*128 + h] = sum_j coef[p,b,i,j] * gl[b,j,h].
// Double-buffered async staging like k_ds_gemm.  grid(8, 32), block 256
__global__ __launch_bounds__(256) void k_xi_gemm(const _Float16* __restrict__ coefh,
                                                 const _Float16* __restrict__ glh,
                                                 float* __restrict__ xi) {
  __shared__ __align__(16) _Float16 As[2][128][LDA_PAD];
  __shared__ __align__(16) _Float16 Bs[2][32][LDB_PAD];
  int bp = blockIdx.y, b = bp >> 2, p = bp & 3;
  const _Float16* Ab = coefh + (size_t)bp * 512 * 512;
  const _Float16* Bb = glh + (size_t)b * 512 * 128;
  int ti = (blockIdx.x >> 1) * 128;  // i tile (4)
  int th = (blockIdx.x & 1) * 64;    // h tile (2)
  int t = threadIdx.x, w = t >> 5, lane = t & 31;
  int mi = (w >> 1) * 32, ni = (w & 1) * 32;
  int rA = t >> 2, cA = (t & 3) * 8;
  int rB = t >> 3, cB = (t & 7) * 8;
  unsigned lA0_0 = ldsoff(&As[0][rA][cA]),      lA0_1 = ldsoff(&As[1][rA][cA]);
  unsigned lA1_0 = ldsoff(&As[0][rA + 64][cA]), lA1_1 = ldsoff(&As[1][rA + 64][cA]);
  unsigned lB_0  = ldsoff(&Bs[0][rB][cB]),      lB_1  = ldsoff(&Bs[1][rB][cB]);
  const _Float16* gA0 = &Ab[(size_t)(ti + rA) * 512 + cA];
  const _Float16* gA1 = &Ab[(size_t)(ti + rA + 64) * 512 + cA];
  const _Float16* gB  = &Bb[(size_t)rB * 128 + th + cB];
  v8f acc00 = {}, acc01 = {}, acc10 = {}, acc11 = {};

  async_issue3(lA0_0, gA0, lA1_0, gA1, lB_0, gB);   // prologue: K-block 0 -> buf 0
  for (int k0 = 0; k0 < 512; k0 += 64) {
    // ---- phase 0: compute buf0 (k0), prefetch k0+32 -> buf1
    if (k0 + 32 < 512) {
      async_issue3(lA0_1, gA0 + k0 + 32, lA1_1, gA1 + k0 + 32,
                   lB_1, gB + (size_t)(k0 + 32) * 128);
      asm volatile("s_wait_asynccnt 0x3" ::: "memory");
    } else {
      asm volatile("s_wait_asynccnt 0x0" ::: "memory");
    }
    __syncthreads();
    {
      v16h a0 = load_a_frag(&As[0][mi][0], LDA_PAD);
      v16h a1 = load_a_frag(&As[0][mi + 16][0], LDA_PAD);
      v16h b0 = load_b_frag(&Bs[0][0][ni], LDB_PAD);
      v16h b1 = load_b_frag(&Bs[0][0][ni + 16], LDB_PAD);
      acc00 = WMMA(a0, b0, acc00);
      acc01 = WMMA(a0, b1, acc01);
      acc10 = WMMA(a1, b0, acc10);
      acc11 = WMMA(a1, b1, acc11);
    }
    __syncthreads();
    // ---- phase 1: compute buf1 (k0+32), prefetch k0+64 -> buf0
    if (k0 + 64 < 512) {
      async_issue3(lA0_0, gA0 + k0 + 64, lA1_0, gA1 + k0 + 64,
                   lB_0, gB + (size_t)(k0 + 64) * 128);
      asm volatile("s_wait_asynccnt 0x3" ::: "memory");
    } else {
      asm volatile("s_wait_asynccnt 0x0" ::: "memory");
    }
    __syncthreads();
    {
      v16h a0 = load_a_frag(&As[1][mi][0], LDA_PAD);
      v16h a1 = load_a_frag(&As[1][mi + 16][0], LDA_PAD);
      v16h b0 = load_b_frag(&Bs[1][0][ni], LDB_PAD);
      v16h b1 = load_b_frag(&Bs[1][0][ni + 16], LDB_PAD);
      acc00 = WMMA(a0, b0, acc00);
      acc01 = WMMA(a0, b1, acc01);
      acc10 = WMMA(a1, b0, acc10);
      acc11 = WMMA(a1, b1, acc11);
    }
    __syncthreads();
  }
  int hh = lane >> 4, nn2 = lane & 15;
  int row0 = ti + mi + 8 * hh;
  int hc0  = th + ni + nn2;
#pragma unroll
  for (int r = 0; r < 8; ++r) {
    size_t o0 = ((size_t)b * 512 + row0 + r) * 512 + p * 128;
    size_t o1 = ((size_t)b * 512 + row0 + 16 + r) * 512 + p * 128;
    xi[o0 + hc0]      = acc00[r];
    xi[o0 + hc0 + 16] = acc01[r];
    xi[o1 + hc0]      = acc10[r];
    xi[o1 + hc0 + 16] = acc11[r];
  }
}

// H = (1 - Z) * Ht   (H0 == 0 collapses the GRU update)
__global__ void k_final(const float* __restrict__ Z, const float* __restrict__ Ht,
                        float* __restrict__ H) {
  int t = blockIdx.x * 256 + threadIdx.x;
  H[t] = (1.f - Z[t]) * Ht[t];
}

// ---------------- Host orchestration ----------------------------------------

struct LP {
  const float *b1, *b2, *b3, *b4, *bl, *w1, *w2, *w3, *w4a, *w4b, *wl;
};
static LP lp_at(void* const* din, int base) {
  LP p;
  p.b1  = (const float*)din[base + 0];
  p.b2  = (const float*)din[base + 1];
  p.b3  = (const float*)din[base + 2];
  p.b4  = (const float*)din[base + 3];
  p.bl  = (const float*)din[base + 4];
  p.w1  = (const float*)din[base + 5];
  p.w2  = (const float*)din[base + 6];
  p.w3  = (const float*)din[base + 7];
  p.w4a = (const float*)din[base + 8];
  p.w4b = (const float*)din[base + 9];
  p.wl  = (const float*)din[base + 10];
  return p;
}

struct WS {
  float *e0, *ne, *xi, *gl, *xA, *xB, *Z, *Ht, *rs, *cs, *ua, *vb, *wa2, *wb3, *cab;
  _Float16 *a2h, *coefh, *glh;
};

static void run_ds(const float* e_in, float* e_out, const WS& w, hipStream_t s) {
  k_rowsum<<<2048, 256, 0, s>>>(e_in, w.rs);
  k_colsum<<<64, 256, 0, s>>>(e_in, w.rs, w.cs);
  k_a2h<<<32768, 256, 0, s>>>(e_in, w.rs, w.cs, w.a2h);
  k_ds_gemm<<<dim3(32, 32), 256, 0, s>>>(w.a2h, e_out);
}

static void run_layer(const float* x, int fin, const LP& lp, const float* e_in,
                      float* ds_out, float* x_out, const WS& w, hipStream_t s) {
  k_gemm_f32<<<dim3(32, 2), 256, 0, s>>>(x, fin, lp.w1, 128, lp.b1, w.gl, 128, fin, 0);
  k_fold<<<4, 128, 0, s>>>(lp.w2, lp.w4a, lp.b2, lp.w3, lp.w4b, lp.b3,
                           w.wa2, w.wb3, w.cab, fin);
  k_uavb<<<64, 256, 0, s>>>(x, fin, w.wa2, w.wb3, w.cab, w.ua, w.vb);
  k_coef<<<32768, 256, 0, s>>>(w.ua, w.vb, lp.b4, e_in ? e_in : w.e0,
                               w.coefh, w.ne, ds_out ? 1 : 0);
  k_f2h<<<2048, 256, 0, s>>>(w.gl, w.glh);
  k_xi_gemm<<<dim3(8, 32), 256, 0, s>>>(w.coefh, w.glh, w.xi);
  k_gemm_f32<<<dim3(32, 2), 256, 0, s>>>(w.xi, 512, lp.wl, 128, lp.bl, x_out, 128, 512, 0);
  if (ds_out) run_ds(w.ne, ds_out, w, s);
}

extern "C" void kernel_launch(void* const* d_in, const int* in_sizes, int n_in,
                              void* d_out, int out_size, void* d_ws, size_t ws_size,
                              hipStream_t stream) {
  (void)in_sizes; (void)n_in; (void)out_size; (void)ws_size;

  // Flattened-pytree input order (sorted dict keys): X, adj, then params
  // {h[0], h[1], lin_h, lin_r, lin_z, r[0], r[1], z[0], z[1]}; layer dicts
  // flatten as {b1,b2,b3,b4,bl,w1,w2,w3,w4a,w4b,wl}.
  const float* X   = (const float*)d_in[0];
  const float* adj = (const float*)d_in[1];
  LP h0 = lp_at(d_in, 2);
  LP h1 = lp_at(d_in, 13);
  const float* wh = (const float*)d_in[24];  // lin_h w [256,128]; rows 0..127 used
  const float* bh = (const float*)d_in[25];
  const float* wz = (const float*)d_in[28];  // lin_z w
  const float* bz = (const float*)d_in[29];
  LP z0 = lp_at(d_in, 52);
  LP z1 = lp_at(d_in, 63);

  char* base = (char*)d_ws;
  size_t off = 0;
  auto carve = [&](size_t bytes) -> void* {
    void* p = base + off;
    off += (bytes + 255) & ~(size_t)255;
    return p;
  };
  WS w;
  w.e0    = (float*)carve(kE * 4);
  w.ne    = (float*)carve(kE * 4);
  w.a2h   = (_Float16*)carve(kE * 2);
  w.coefh = (_Float16*)carve(kE * 2);
  w.xi    = (float*)carve((size_t)kBN * 512 * 4);
  w.gl    = (float*)carve((size_t)kBN * 128 * 4);
  w.glh   = (_Float16*)carve((size_t)kBN * 128 * 2);
  w.xA    = (float*)carve((size_t)kBN * 128 * 4);
  w.xB    = (float*)carve((size_t)kBN * 128 * 4);
  w.Z     = (float*)carve((size_t)kBN * 128 * 4);
  w.Ht    = (float*)carve((size_t)kBN * 128 * 4);
  w.rs    = (float*)carve(16384 * 4);
  w.cs    = (float*)carve(16384 * 4);
  w.ua    = (float*)carve(16384 * 4);
  w.vb    = (float*)carve(16384 * 4);
  w.wa2   = (float*)carve(512 * 4);
  w.wb3   = (float*)carve(512 * 4);
  w.cab   = (float*)carve(8 * 4);

  float* H = (float*)d_out;                 // [B,N,128]
  float* E = H + (size_t)kBN * 128;         // [B,N,N,P]

  // Shared initial Sinkhorn pass
  run_ds(adj, w.e0, w, stream);

  // z branch (carries the E chain; final _ds writes E directly to d_out)
  run_layer(X,    64,  z0, w.e0, w.e0, w.xA, w, stream);
  run_layer(w.xA, 128, z1, w.e0, E,    w.xB, w, stream);
  k_gemm_f32<<<dim3(32, 2), 256, 0, stream>>>(w.xB, 128, wz, 128, bz, w.Z, 128, 128, 1);

  // h branch (its e-chain is dead since H0 == 0: skip all _ds work)
  run_layer(X,    64,  h0, nullptr, nullptr, w.xA, w, stream);
  run_layer(w.xA, 128, h1, nullptr, nullptr, w.xB, w, stream);
  k_gemm_f32<<<dim3(32, 2), 256, 0, stream>>>(w.xB, 128, wh, 128, bh, w.Ht, 128, 128, 2);

  // H = (1 - Z) * Ht   (r branch is dead code: R only scales H0 == 0)
  k_final<<<2048, 256, 0, stream>>>(w.Z, w.Ht, H);
}
